// GATwithLSTM_74337293959508
// MI455X (gfx1250) — matrix operand order repaired and checked by reading, not compiled, not run on hardware
//
#include <hip/hip_runtime.h>

// ---------------------------------------------------------------------------
// GAT(2 layers) + LSTM + projection for MI455X (gfx1250), wave32, WMMA f16.
// ---------------------------------------------------------------------------

typedef _Float16 f16_t;
typedef __attribute__((ext_vector_type(16))) _Float16 v16h;
typedef __attribute__((ext_vector_type(8)))  _Float16 v8h;
typedef __attribute__((ext_vector_type(8)))  float    v8f;

#define CDIV(a, b) (((a) + (b) - 1) / (b))

#define CAT16(lo, hi) \
  __builtin_shufflevector(lo, hi, 0, 1, 2, 3, 4, 5, 6, 7, 8, 9, 10, 11, 12, 13, 14, 15)

// ------------------------------- utility kernels ---------------------------

__global__ void k_zero_u32(unsigned* __restrict__ p, long n) {
  long i = (long)blockIdx.x * blockDim.x + threadIdx.x;
  long s = (long)gridDim.x * blockDim.x;
  for (; i < n; i += s) p[i] = 0u;
}

__global__ void k_gather(const int* __restrict__ ids, const float* __restrict__ emb,
                         f16_t* __restrict__ x, long total) {
  long i = (long)blockIdx.x * 256 + threadIdx.x;
  if (i >= total) return;
  long n = i >> 8;
  int  c = (int)(i & 255);
  int id = ids[n];
  float v = (id == 0) ? 0.f : emb[(long)id * 256 + c];   // emb.at[0].set(0)
  x[i] = (f16_t)v;
}

__global__ void k_cvt(const float* __restrict__ in, f16_t* __restrict__ out, long n) {
  long i = (long)blockIdx.x * 256 + threadIdx.x;
  if (i < n) out[i] = (f16_t)in[i];
}

// out[j][k] = in[k][j]  (output-linear index; K is a power of two)
__global__ void k_cvtT(const float* __restrict__ in, f16_t* __restrict__ out,
                       int kshift, int kmask, int ncols, long total) {
  long i = (long)blockIdx.x * 256 + threadIdx.x;
  if (i >= total) return;
  long j = i >> kshift;
  int  k = (int)(i & kmask);
  out[i] = (f16_t)in[(long)k * ncols + j];
}

__global__ void k_biasg(const float* __restrict__ a, const float* __restrict__ b,
                        float* __restrict__ o, int n) {
  int i = blockIdx.x * 256 + threadIdx.x;
  if (i < n) o[i] = a[i] + b[i];
}

// out_f16 = (elu?)(in + bias[c])
__global__ void k_bias_act(const float* __restrict__ in, const float* __restrict__ bias,
                           f16_t* __restrict__ out, long total, int cmask, int elu) {
  long i = (long)blockIdx.x * 256 + threadIdx.x;
  if (i >= total) return;
  float v = in[i] + bias[i & cmask];
  if (elu) v = (v > 0.f) ? v : (__expf(v) - 1.f);
  out[i] = (f16_t)v;
}

// ------------------------------- WMMA GEMM ---------------------------------
// C[M][N] = A[M][K] * Bt[N][K]^T (+bias[N]).  f16 in, f32 accumulate.
// Block tile 128x64, 256 threads (8 waves), each wave: 2x2 of 16x16 tiles.
// LDS rows padded to 80B (16B-aligned, conflict-free b128 fragment loads).

#define TM 128
#define TN 64
#define TK 32
#define APAD 40   // padded K stride in LDS (halves); 80B rows

__global__ __launch_bounds__(256) void k_gemm_f16(
    const f16_t* __restrict__ A,   // [M][K]
    const f16_t* __restrict__ Bt,  // [N][K]
    const float* __restrict__ bias,// [N] or nullptr
    float* __restrict__ Cf,        // [M][N] or nullptr
    f16_t* __restrict__ Ch,        // [M][N] or nullptr
    int M, int N, int K) {
  __shared__ alignas(16) f16_t As[TM][APAD];
  __shared__ alignas(16) f16_t Bs[TN][APAD];
  const int m0 = blockIdx.x * TM;
  const int n0 = blockIdx.y * TN;
  const int tid = threadIdx.x;
  const int lane = tid & 31;
  const int w = tid >> 5;
  const int wm = w >> 1;   // 0..3
  const int wn = w & 1;    // 0..1

  v8f acc[2][2] = {};

  for (int k0 = 0; k0 < K; k0 += TK) {
    __syncthreads();
    {  // A: 128 rows x 32 halves; thread -> row tid>>1, 16-half chunk (2x b128)
      int r = tid >> 1, c = (tid & 1) * 16;
      int gm = m0 + r;
      v8h a0 = {}, a1 = {};
      if (gm < M) {
        const v8h* sp = (const v8h*)(A + (long)gm * K + k0 + c);
        a0 = sp[0];
        a1 = sp[1];
      }
      *(v8h*)&As[r][c]     = a0;
      *(v8h*)&As[r][c + 8] = a1;
    }
    {  // B: 64 rows x 32 halves; thread -> row tid>>2, 8-half chunk (1x b128)
      int r = tid >> 2, c = (tid & 3) * 8;
      int gn = n0 + r;
      v8h b0 = {};
      if (gn < N) b0 = *(const v8h*)(Bt + (long)gn * K + k0 + c);
      *(v8h*)&Bs[r][c] = b0;
    }
    __syncthreads();

    // A fragments: lane<16 -> K {0..7,16..23}; lane>=16 -> K {8..15,24..31}
    v16h af[2], bf[2];
    const int am = lane & 15;
    const int kb = (lane >> 4) * 8;
#pragma unroll
    for (int i = 0; i < 2; ++i) {
      int mr = wm * 32 + i * 16 + am;
      v8h lo = *(const v8h*)&As[mr][kb];
      v8h hi = *(const v8h*)&As[mr][kb + 16];
      af[i] = CAT16(lo, hi);
    }
    // B fragments: lane<16 -> K {0..7,16..23}; lane>=16 -> K {8..15,24..31}
    const int kbb = (lane >> 4) * 8;
#pragma unroll
    for (int j = 0; j < 2; ++j) {
      int nr = wn * 32 + j * 16 + am;
      v8h b0 = *(const v8h*)&Bs[nr][kbb];
      v8h b1 = *(const v8h*)&Bs[nr][kbb + 16];
      bf[j] = CAT16(b0, b1);
    }
#pragma unroll
    for (int i = 0; i < 2; ++i)
#pragma unroll
      for (int j = 0; j < 2; ++j)
        acc[i][j] = __builtin_amdgcn_wmma_f32_16x16x32_f16(
            false, af[i], false, bf[j], (short)0, acc[i][j], false, false);
  }

  // Epilogue: D layout m = v + 8*(lane>>4), n = lane&15
#pragma unroll
  for (int i = 0; i < 2; ++i)
#pragma unroll
    for (int j = 0; j < 2; ++j) {
      int mb = m0 + wm * 32 + i * 16 + (lane >> 4) * 8;
      int n = n0 + wn * 32 + j * 16 + (lane & 15);
      if (n < N) {
        float bv = bias ? bias[n] : 0.f;
#pragma unroll
        for (int v = 0; v < 8; ++v) {
          int m = mb + v;
          if (m < M) {
            float val = acc[i][j][v] + bv;
            if (Cf) Cf[(long)m * N + n] = val;
            if (Ch) Ch[(long)m * N + n] = (f16_t)val;
          }
        }
      }
    }
}

// ------------------------------- attention ---------------------------------

// alpha_s[n][h] = sum_d h[n][h][d]*a_src[h][d]  (one wave per head)
__global__ void k_alpha(const f16_t* __restrict__ h, const float* __restrict__ a_src,
                        const float* __restrict__ a_dst, float* __restrict__ as_,
                        float* __restrict__ ad_, int heads) {
  int n = blockIdx.x;
  int head = threadIdx.x >> 5;
  int lane = threadIdx.x & 31;
  const f16_t* hp = h + ((long)n * heads + head) * 256;
  const float* sp = a_src + head * 256;
  const float* dp = a_dst + head * 256;
  float ss = 0.f, sd = 0.f;
  for (int i = lane; i < 256; i += 32) {
    float hv = (float)hp[i];
    ss += hv * sp[i];
    sd += hv * dp[i];
  }
  for (int off = 16; off > 0; off >>= 1) {
    ss += __shfl_down(ss, off, 32);
    sd += __shfl_down(sd, off, 32);
  }
  if (lane == 0) {
    as_[n * heads + head] = ss;
    ad_[n * heads + head] = sd;
  }
}

__device__ __forceinline__ unsigned f2key(float f) {
  unsigned u = __float_as_uint(f);
  return (u & 0x80000000u) ? ~u : (u | 0x80000000u);
}
__device__ __forceinline__ float key2f(unsigned k) {
  unsigned u = (k & 0x80000000u) ? (k & 0x7FFFFFFFu) : ~k;
  return __uint_as_float(u);
}

// e = leaky_relu(as[src]+ad[dst], 0.2); segment max via atomicMax on uint keys
__global__ void k_edge_e(const float* __restrict__ as_, const float* __restrict__ ad_,
                         const int* __restrict__ ei, int E, long total, int heads,
                         int hshift, float* __restrict__ e_out,
                         unsigned* __restrict__ mkeys) {
  long idx = (long)blockIdx.x * 256 + threadIdx.x;
  if (idx >= total) return;
  int hh = (int)(idx & (heads - 1));
  long e = idx >> hshift;
  int s, d;
  if (e < E) { s = ei[e]; d = ei[E + e]; } else { s = d = (int)(e - E); }
  float v = as_[(long)s * heads + hh] + ad_[(long)d * heads + hh];
  v = (v > 0.f) ? v : 0.2f * v;
  e_out[idx] = v;
  atomicMax(&mkeys[(long)d * heads + hh], f2key(v));
}

// p = exp(e - m[dst]); z[dst] += p
__global__ void k_edge_p(float* __restrict__ e_inout, const unsigned* __restrict__ mkeys,
                         float* __restrict__ z, const int* __restrict__ ei, int E,
                         long total, int heads, int hshift) {
  long idx = (long)blockIdx.x * 256 + threadIdx.x;
  if (idx >= total) return;
  int hh = (int)(idx & (heads - 1));
  long e = idx >> hshift;
  int d;
  if (e < E) d = ei[E + e]; else d = (int)(e - E);
  float p = __expf(e_inout[idx] - key2f(mkeys[(long)d * heads + hh]));
  e_inout[idx] = p;
  atomicAdd(&z[(long)d * heads + hh], p);
}

// out[dst][h][c] += (p/(z[dst][h]+1e-16)) * hsrc[src][h][c]
__global__ void k_edge_agg(const float* __restrict__ p, const float* __restrict__ z,
                           const int* __restrict__ ei, int E, long total, int heads,
                           int hshift, const f16_t* __restrict__ hsrc,
                           float* __restrict__ out) {
  long idx = (long)blockIdx.x * 256 + threadIdx.x;
  if (idx >= total) return;
  int c = (int)(idx & 255);
  long eh = idx >> 8;
  int hh = (int)(eh & (heads - 1));
  long e = eh >> hshift;
  int s, d;
  if (e < E) { s = ei[e]; d = ei[E + e]; } else { s = d = (int)(e - E); }
  float alpha = p[eh] / (z[(long)d * heads + hh] + 1e-16f);
  float val = alpha * (float)hsrc[((long)s * heads + hh) * 256 + c];
  atomicAdd(&out[((long)d * heads + hh) * 256 + c], val);
}

// ------------------------------- LSTM --------------------------------------
// 16 blocks x 256 threads. Block g owns batch rows [16g,16g+16).
// Per step: G(16x1024) = h_f16(16x256) @ Whh^T via WMMA, + precomputed Xg
// (input GEMM + bih + bhh), then fused gate update; c lives in registers, h in LDS.

__global__ __launch_bounds__(256) void k_lstm(const float* __restrict__ Xg,   // [B*64][1024]
                                              const f16_t* __restrict__ Whh,  // [1024][256]
                                              f16_t* __restrict__ hT_h) {     // [B][256]
  __shared__ alignas(16) f16_t hs[16][256];
  __shared__ float Gs[16][1024];
  const int tid = threadIdx.x;
  const int lane = tid & 31;
  const int w = tid >> 5;     // 8 waves; wave covers N-cols [w*128, w*128+128)
  const int g = blockIdx.x;   // 0..15
  const int sm = tid >> 4;    // state row 0..15
  const int sj = tid & 15;    // state col base

  for (int i = tid; i < 16 * 256; i += 256) ((f16_t*)hs)[i] = (f16_t)0.f;
  float cst[16];
#pragma unroll
  for (int q = 0; q < 16; ++q) cst[q] = 0.f;
  __syncthreads();

  const int am = lane & 15;
  for (int t = 0; t < 64; ++t) {
    v8f acc[8] = {};
#pragma unroll
    for (int ks = 0; ks < 8; ++ks) {  // K = 256
      int kb = ks * 32 + (lane >> 4) * 8;
      v8h lo = *(const v8h*)&hs[am][kb];
      v8h hi = *(const v8h*)&hs[am][kb + 16];
      v16h af = CAT16(lo, hi);
#pragma unroll
      for (int s = 0; s < 8; ++s) {
        int j = w * 128 + s * 16 + am;
        const v8h* bp =
            (const v8h*)(Whh + (long)j * 256 + ks * 32 + (lane >> 4) * 8);
        v16h bf = CAT16(bp[0], bp[2]);
        acc[s] = __builtin_amdgcn_wmma_f32_16x16x32_f16(
            false, af, false, bf, (short)0, acc[s], false, false);
      }
    }
    // dump accumulators: m = v + 8*(lane>>4), n = lane&15
#pragma unroll
    for (int s = 0; s < 8; ++s) {
      int n = w * 128 + s * 16 + am;
      int mb = (lane >> 4) * 8;
#pragma unroll
      for (int v = 0; v < 8; ++v) Gs[mb + v][n] = acc[s][v];
    }
    __syncthreads();
    // gates
    const int b = g * 16 + sm;
    const float* xrow = Xg + ((long)b * 64 + t) * 1024;
#pragma unroll
    for (int q = 0; q < 16; ++q) {
      int j = sj + q * 16;
      float gi = Gs[sm][j]       + xrow[j];
      float gf = Gs[sm][256 + j] + xrow[256 + j];
      float gg = Gs[sm][512 + j] + xrow[512 + j];
      float go = Gs[sm][768 + j] + xrow[768 + j];
      float si = 1.f / (1.f + __expf(-gi));
      float sf = 1.f / (1.f + __expf(-gf));
      float so = 1.f / (1.f + __expf(-go));
      float cn = sf * cst[q] + si * tanhf(gg);
      cst[q] = cn;
      float hn = so * tanhf(cn);
      hs[sm][j] = (f16_t)hn;
      if (t == 63) hT_h[(long)b * 256 + j] = (f16_t)hn;
    }
    __syncthreads();
  }
}

// ------------------------------- driver ------------------------------------

extern "C" void kernel_launch(void* const* d_in, const int* in_sizes, int n_in,
                              void* d_out, int out_size, void* d_ws, size_t ws_size,
                              hipStream_t stream) {
  const int kN = 16384, kE = 131072, kEtot = kE + kN;

  const int*   x_ids = (const int*)d_in[0];
  const int*   ei    = (const int*)d_in[1];   // [2][E]
  const float* emb   = (const float*)d_in[3];
  const float* W1    = (const float*)d_in[4];
  const float* aS1   = (const float*)d_in[5];
  const float* aD1   = (const float*)d_in[6];
  const float* b1    = (const float*)d_in[7];
  const float* W2    = (const float*)d_in[8];
  const float* aS2   = (const float*)d_in[9];
  const float* aD2   = (const float*)d_in[10];
  const float* b2    = (const float*)d_in[11];
  const float* Wih   = (const float*)d_in[12];
  const float* Whh   = (const float*)d_in[13];
  const float* bih   = (const float*)d_in[14];
  const float* bhh   = (const float*)d_in[15];
  const float* Wp    = (const float*)d_in[16];
  const float* bp    = (const float*)d_in[17];
  float* out = (float*)d_out;

  // ---- workspace bump allocation (with explicit dead-buffer aliasing) ----
  char* cur = (char*)d_ws;
  auto alloc = [&](size_t bytes) {
    char* r = cur;
    cur += (bytes + 255) & ~(size_t)255;
    return r;
  };
  f16_t*    xh   = (f16_t*)alloc((size_t)kN * 256 * 2);       // x f16; later x3
  f16_t*    Wt1  = (f16_t*)alloc((size_t)2048 * 256 * 2);     // W1^T f16; later Wt2
  f16_t*    h1   = (f16_t*)alloc((size_t)kN * 2048 * 2);      // later Xg (f32 Nx1024)
  float*    as1  = (float*)alloc((size_t)kN * 8 * 4);
  float*    ad1  = (float*)alloc((size_t)kN * 8 * 4);
  unsigned* mk1  = (unsigned*)alloc((size_t)kN * 8 * 4);      // contiguous with z1
  float*    z1   = (float*)alloc((size_t)kN * 8 * 4);
  float*    p1   = (float*)alloc((size_t)kEtot * 8 * 4);
  float*    out1 = (float*)alloc((size_t)kN * 2048 * 4);
  f16_t*    x2   = (f16_t*)alloc((size_t)kN * 2048 * 2);      // later Wp f16
  f16_t*    h2   = (f16_t*)alloc((size_t)kN * 256 * 2);
  float*    as2  = (float*)alloc((size_t)kN * 4);
  float*    ad2  = (float*)alloc((size_t)kN * 4);
  unsigned* mk2  = (unsigned*)alloc((size_t)kN * 4);          // contiguous with z2
  float*    z2   = (float*)alloc((size_t)kN * 4);
  float*    p2   = (float*)alloc((size_t)kEtot * 4);
  float*    out2 = (float*)alloc((size_t)kN * 256 * 4);
  f16_t*    Wihh = (f16_t*)alloc((size_t)1024 * 256 * 2);
  f16_t*    Whhh = (f16_t*)alloc((size_t)1024 * 256 * 2);
  float*    bg   = (float*)alloc((size_t)1024 * 4);
  f16_t*    hTh  = (f16_t*)alloc((size_t)256 * 256 * 2);
  // aliases of dead regions
  f16_t* x3h = xh;                 // x dead after GEMM1
  f16_t* Wt2 = Wt1;                // Wt1 dead after GEMM1
  float* Xg  = (float*)h1;         // h1 dead after layer-1 aggregation (same 64MB)
  f16_t* Wph = x2;                 // x2 dead after GEMM2

  // ---- zero accumulators / softmax state ----
  k_zero_u32<<<2048, 256, 0, stream>>>((unsigned*)mk1, (long)kN * 8 * 2);     // mk1+z1
  k_zero_u32<<<4096, 256, 0, stream>>>((unsigned*)out1, (long)kN * 2048);
  k_zero_u32<<<256, 256, 0, stream>>>((unsigned*)mk2, (long)kN * 2);          // mk2+z2
  k_zero_u32<<<2048, 256, 0, stream>>>((unsigned*)out2, (long)kN * 256);

  // ---- embedding gather + weight conversion ----
  long nx = (long)kN * 256;
  k_gather<<<CDIV(nx, 256), 256, 0, stream>>>(x_ids, emb, xh, nx);
  k_cvtT<<<CDIV(2048L * 256, 256), 256, 0, stream>>>(W1, Wt1, 8, 255, 2048,
                                                     2048L * 256);

  // ---- GAT layer 1 ----
  {
    dim3 grid(CDIV(kN, TM), CDIV(2048, TN));
    k_gemm_f16<<<grid, 256, 0, stream>>>(xh, Wt1, nullptr, nullptr, h1,
                                         kN, 2048, 256);
  }
  k_alpha<<<kN, 256, 0, stream>>>(h1, aS1, aD1, as1, ad1, 8);
  {
    long tot = (long)kEtot * 8;
    k_edge_e<<<CDIV(tot, 256), 256, 0, stream>>>(as1, ad1, ei, kE, tot, 8, 3, p1, mk1);
    k_edge_p<<<CDIV(tot, 256), 256, 0, stream>>>(p1, mk1, z1, ei, kE, tot, 8, 3);
    long tota = tot * 256;
    k_edge_agg<<<CDIV(tota, 256), 256, 0, stream>>>(p1, z1, ei, kE, tota, 8, 3,
                                                    h1, out1);
  }
  // x2 = f16(elu(out1 + b1))
  k_bias_act<<<CDIV((long)kN * 2048, 256), 256, 0, stream>>>(
      out1, b1, x2, (long)kN * 2048, 2047, 1);

  // ---- GAT layer 2 ----
  k_cvtT<<<CDIV(256L * 2048, 256), 256, 0, stream>>>(W2, Wt2, 11, 2047, 256,
                                                     256L * 2048);
  {
    dim3 grid(CDIV(kN, TM), CDIV(256, TN));
    k_gemm_f16<<<grid, 256, 0, stream>>>(x2, Wt2, nullptr, nullptr, h2,
                                         kN, 256, 2048);
  }
  k_alpha<<<kN, 32, 0, stream>>>(h2, aS2, aD2, as2, ad2, 1);
  {
    long tot = (long)kEtot;
    k_edge_e<<<CDIV(tot, 256), 256, 0, stream>>>(as2, ad2, ei, kE, tot, 1, 0, p2, mk2);
    k_edge_p<<<CDIV(tot, 256), 256, 0, stream>>>(p2, mk2, z2, ei, kE, tot, 1, 0);
    long tota = tot * 256;
    k_edge_agg<<<CDIV(tota, 256), 256, 0, stream>>>(p2, z2, ei, kE, tota, 1, 0,
                                                    h2, out2);
  }
  // x3 = f16(out2 + b2)
  k_bias_act<<<CDIV((long)kN * 256, 256), 256, 0, stream>>>(
      out2, b2, x3h, (long)kN * 256, 255, 0);

  // ---- LSTM: input pre-GEMM (bias bih+bhh folded), then WMMA recurrence ----
  k_cvt<<<CDIV(1024L * 256, 256), 256, 0, stream>>>(Wih, Wihh, 1024L * 256);
  k_cvt<<<CDIV(1024L * 256, 256), 256, 0, stream>>>(Whh, Whhh, 1024L * 256);
  k_biasg<<<4, 256, 0, stream>>>(bih, bhh, bg, 1024);
  {
    dim3 grid(CDIV(kN, TM), CDIV(1024, TN));
    k_gemm_f16<<<grid, 256, 0, stream>>>(x3h, Wihh, bg, Xg, nullptr,
                                         kN, 1024, 256);
  }
  k_lstm<<<16, 256, 0, stream>>>(Xg, Whhh, hTh);

  // ---- final projection: out = hT @ Wp^T + bp ----
  k_cvt<<<CDIV(37000L * 256, 256), 256, 0, stream>>>(Wp, Wph, 37000L * 256);
  {
    dim3 grid(CDIV(256, TM), CDIV(37000, TN));
    k_gemm_f16<<<grid, 256, 0, stream>>>(hTh, Wph, bp, out, nullptr,
                                         256, 37000, 256);
  }
  (void)in_sizes; (void)n_in; (void)out_size; (void)ws_size;
}
